// Block_49624052138029
// MI455X (gfx1250) — matrix operand order repaired and checked by reading, not compiled, not run on hardware
//
#include <hip/hip_runtime.h>
#include <hip/hip_bf16.h>
#include <math.h>

typedef __bf16 bf16;
typedef __attribute__((ext_vector_type(16))) __bf16 v16bf;
typedef __attribute__((ext_vector_type(8)))  float  v8f;
typedef __attribute__((address_space(3))) bf16 lds_bf16;

#define D_MODEL 1024
#define NHEADS  16
#define HDIM    64
#define SEQ     2048
#define BATCH   2
#define MTOK    (BATCH*SEQ)          // 4096 rows
// softmax in exp2 domain: fold d_model^-0.5 * log2(e) into Q once
#define QSCALE_L2E 0.045084171507297876f

union Frag { v16bf v; uint4 q[2]; };

// ---- WMMA wrapper: D = A(16x32 bf16) * B(32x16 bf16) + C (f32) ----
__device__ inline v8f wmma_bf16(v16bf a, v16bf b, v8f c) {
  return __builtin_amdgcn_wmma_f32_16x16x32_bf16(false, a, false, b, (short)0, c, false, false);
}

// A fragment 16x32: lane(0-15) row=lane, K {0..7,16..23}; lane(16-31) row=lane-16, K {8..15,24..31}
__device__ inline v16bf load_frag_A(const bf16* base, int stride_h) {
  int lane = threadIdx.x & 31;
  const bf16* p = base + (size_t)(lane & 15) * stride_h + (lane >> 4) * 8;
  Frag f;
  f.q[0] = *(const uint4*)(p);
  f.q[1] = *(const uint4*)(p + 16);
  return f.v;
}

// B fragment 32x16 from B^T layout [16 cols][K...]: lane(0-15) col=lane K {0..7,16..23}; lane(16-31) col=lane-16 K {8..15,24..31}
__device__ inline v16bf load_frag_B(const bf16* base, int stride_h) {
  int lane = threadIdx.x & 31;
  const bf16* p = base + (size_t)(lane & 15) * stride_h + (lane >> 4) * 8;
  Frag f;
  f.q[0] = *(const uint4*)(p);
  f.q[1] = *(const uint4*)(p + 16);
  return f.v;
}

// ---- CDNA5 async global->LDS copy (ASYNCcnt) ----
__device__ inline uint32_t lds_off(const bf16* p) {
  return (uint32_t)(uintptr_t)(lds_bf16*)p;
}
__device__ inline void async_b128(uint32_t lds, const bf16* g) {
  asm volatile("global_load_async_to_lds_b128 %0, %1, off"
               :: "v"(lds), "v"((unsigned long long)(uintptr_t)g) : "memory");
}
__device__ inline void wait_async0() {
  asm volatile("s_wait_asynccnt 0x0" ::: "memory");
}

// ---- fused DPP cross-lane reductions over 16-lane halves ----
// one instruction per hop: vdst = op(dpp8(src), src); xor8 hop via ds_swizzle SWAPX8
__device__ inline float swz_x8(float x) {
  return __int_as_float(__builtin_amdgcn_ds_swizzle(__float_as_int(x), 0x201F));
}
__device__ inline float red_max16(float v) {
  float t;
  asm("v_max_num_f32_dpp %0, %1, %1 dpp8:[1,0,3,2,5,4,7,6]" : "=v"(t) : "v"(v));
  asm("v_max_num_f32_dpp %0, %1, %1 dpp8:[2,3,0,1,6,7,4,5]" : "=v"(v) : "v"(t));
  asm("v_max_num_f32_dpp %0, %1, %1 dpp8:[4,5,6,7,0,1,2,3]" : "=v"(t) : "v"(v));
  return fmaxf(t, swz_x8(t));
}
__device__ inline float red_sum16(float v) {
  float t;
  asm("v_add_f32_dpp %0, %1, %1 dpp8:[1,0,3,2,5,4,7,6]" : "=v"(t) : "v"(v));
  asm("v_add_f32_dpp %0, %1, %1 dpp8:[2,3,0,1,6,7,4,5]" : "=v"(v) : "v"(t));
  asm("v_add_f32_dpp %0, %1, %1 dpp8:[4,5,6,7,0,1,2,3]" : "=v"(t) : "v"(v));
  return t + swz_x8(t);
}

// ---------------- weight convert / transpose ----------------
__global__ __launch_bounds__(256) void k_cvt_qkv(const float* __restrict__ Wq,
                                                 const float* __restrict__ Wk,
                                                 const float* __restrict__ Wv,
                                                 bf16* __restrict__ out) {
  int o = blockIdx.x * 256 + threadIdx.x;
  if (o >= 3072 * 1024) return;
  int col = o >> 10, k = o & 1023;
  int which = col >> 10, h = (col >> 6) & 15, e = col & 63;
  const float* W = (which == 0) ? Wq : (which == 1) ? Wk : Wv;
  out[o] = (bf16)W[((size_t)h * 1024 + k) * 64 + e];
}

__global__ __launch_bounds__(256) void k_cvt_T(const float* __restrict__ W,
                                               bf16* __restrict__ out) {
  int o = blockIdx.x * 256 + threadIdx.x;
  int n = o >> 10, k = o & 1023;
  out[o] = (bf16)W[(size_t)k * 1024 + n];
}

// ---------------- layernorm: fp32 in -> bf16 out ----------------
__global__ __launch_bounds__(256) void k_ln(const float* __restrict__ X,
                                            const float* __restrict__ G,
                                            const float* __restrict__ Bt,
                                            bf16* __restrict__ H) {
  __shared__ float red[16];
  int row = blockIdx.x, tid = threadIdx.x;
  const float* x = X + (size_t)row * D_MODEL;
  float v[4], s = 0.f, s2 = 0.f;
#pragma unroll
  for (int j = 0; j < 4; j++) { v[j] = x[tid + j * 256]; s += v[j]; s2 += v[j] * v[j]; }
#pragma unroll
  for (int off = 16; off; off >>= 1) { s += __shfl_xor(s, off, 32); s2 += __shfl_xor(s2, off, 32); }
  int wave = tid >> 5, lane = tid & 31;
  if (lane == 0) { red[wave] = s; red[8 + wave] = s2; }
  __syncthreads();
  float ts = 0.f, ts2 = 0.f;
#pragma unroll
  for (int w = 0; w < 8; w++) { ts += red[w]; ts2 += red[8 + w]; }
  float mu = ts * (1.f / D_MODEL);
  float var = ts2 * (1.f / D_MODEL) - mu * mu;
  float rs = rsqrtf(var + 1e-5f);
  bf16* h = H + (size_t)row * D_MODEL;
#pragma unroll
  for (int j = 0; j < 4; j++) {
    int i = tid + j * 256;
    h[i] = (bf16)((v[j] - mu) * rs * G[i] + Bt[i]);
  }
}

// ---------------- bf16 GEMM: C = A[M,K] * BT[N,K]^T, async double-buffered ----
// workgroup tile 256x64, wave tile 32x64, K-step 32
// MODE 0: scatter bf16 -> Q (scaled) / K / V^T buffers
// MODE 1/3: fp32 out = resid + acc + bias
// MODE 2: bf16 out = gelu(acc + bias)
#define TM 256
#define TN 64
#define TK 32
#define LDP 40
#define ASZ (TM*LDP)
#define BSZ (TN*LDP)

template <int MODE>
__global__ __launch_bounds__(256) void k_gemm(const bf16* __restrict__ A,
                                              const bf16* __restrict__ BT,
                                              const float* __restrict__ bias,
                                              const float* __restrict__ resid,
                                              float* __restrict__ outF,
                                              bf16* __restrict__ outH,
                                              bf16* __restrict__ qb,
                                              bf16* __restrict__ kb,
                                              bf16* __restrict__ vb,
                                              int N, int K) {
  __shared__ __align__(16) bf16 As[2 * ASZ];
  __shared__ __align__(16) bf16 Bs[2 * BSZ];
  int tid = threadIdx.x;
  int wave = tid >> 5, lane = tid & 31;
  int m0 = blockIdx.y * TM, n0 = blockIdx.x * TN;

  v8f acc[2][4];
#pragma unroll
  for (int s = 0; s < 2; s++)
#pragma unroll
    for (int c = 0; c < 4; c++)
#pragma unroll
      for (int g = 0; g < 8; g++) acc[s][c][g] = 0.f;

  int ar = tid >> 1, ao = (tid & 1) * 16;   // A: 2 rows x 16 halves / thread
  int br = tid >> 2, bo2 = (tid & 3) * 8;   // B: 8 halves / thread

  auto issue = [&](int p, int kb0) {
    const bf16* g0 = A + (size_t)(m0 + ar) * K + kb0 + ao;
    uint32_t l0 = lds_off(&As[p * ASZ + ar * LDP + ao]);
    async_b128(l0, g0);
    async_b128(l0 + 16, g0 + 8);
    const bf16* g1 = A + (size_t)(m0 + ar + 128) * K + kb0 + ao;
    uint32_t l1 = lds_off(&As[p * ASZ + (ar + 128) * LDP + ao]);
    async_b128(l1, g1);
    async_b128(l1 + 16, g1 + 8);
    async_b128(lds_off(&Bs[p * BSZ + br * LDP + bo2]),
               BT + (size_t)(n0 + br) * K + kb0 + bo2);
  };

  issue(0, 0);
  int p = 0;
  for (int kb0 = 0; kb0 < K; kb0 += TK) {
    wait_async0();
    __syncthreads();
    if (kb0 + TK < K) issue(p ^ 1, kb0 + TK);  // stream next slab into other buffer
    const bf16* Ab = &As[p * ASZ + (wave * 32) * LDP];
    v16bf a0 = load_frag_A(Ab, LDP);
    v16bf a1 = load_frag_A(Ab + 16 * LDP, LDP);
#pragma unroll
    for (int ct = 0; ct < 4; ct++) {
      v16bf b = load_frag_B(&Bs[p * BSZ + (ct * 16) * LDP], LDP);
      acc[0][ct] = wmma_bf16(a0, b, acc[0][ct]);
      acc[1][ct] = wmma_bf16(a1, b, acc[1][ct]);
    }
    p ^= 1;
  }

  int fh = lane >> 4, fr = lane & 15;
#pragma unroll
  for (int sub = 0; sub < 2; sub++) {
#pragma unroll
    for (int ct = 0; ct < 4; ct++) {
#pragma unroll
      for (int g = 0; g < 8; g++) {
        int row = m0 + wave * 32 + sub * 16 + g + fh * 8;
        int col = n0 + ct * 16 + fr;
        float v = acc[sub][ct][g];
        if (MODE == 0) {
          int b = row >> 11, t = row & 2047;
          int which = col >> 10, h = (col >> 6) & 15, e = col & 63;
          if (which == 0) {  // Q: fold softmax scale * log2(e)
            qb[(((size_t)b * NHEADS + h) * SEQ + t) * HDIM + e] = (bf16)(v * QSCALE_L2E);
          } else if (which == 1) {
            kb[(((size_t)b * NHEADS + h) * SEQ + t) * HDIM + e] = (bf16)v;
          } else {  // V stored transposed: [b,h,e,t] so attention V tile is a plain copy
            vb[(((size_t)b * NHEADS + h) * HDIM + e) * SEQ + t] = (bf16)v;
          }
        } else if (MODE == 1 || MODE == 3) {
          outF[(size_t)row * N + col] = resid[(size_t)row * N + col] + v + bias[col];
        } else {
          float u = v + bias[col];
          float gel = 0.5f * u * (1.f + erff(u * 0.70710678118654752f));
          outH[(size_t)row * N + col] = (bf16)gel;
        }
      }
    }
  }
}

// ---------------- flash attention: 64 keys / step, 16 WMMA / step ----------------
// grid (SEQ/128, B*H), 8 waves, wave = 16 query rows; all staging via async->LDS
__global__ __launch_bounds__(256) void k_attn(const bf16* __restrict__ Q,
                                              const bf16* __restrict__ Kb,
                                              const bf16* __restrict__ Vt,
                                              bf16* __restrict__ O) {
  __shared__ __align__(16) bf16 Ks[64 * 72];        // [key][dim], stride 72
  __shared__ __align__(16) bf16 Vs[64 * 72];        // V^T: [dim][key], stride 72
  __shared__ __align__(16) bf16 Ps[8 * 16 * 72];    // per-wave P tile [16][64], stride 72
  int tid = threadIdx.x, wave = tid >> 5, lane = tid & 31;
  int fh = lane >> 4, fr = lane & 15;
  int bh = blockIdx.y;
  size_t base = (size_t)bh * SEQ * HDIM;            // same size for [t][e] and [e][t]
  int q0 = blockIdx.x * 128 + wave * 16;

  // Q fragments (Q pre-scaled by QSCALE_L2E in the QKV epilogue)
  const bf16* qrow = Q + base + (size_t)(q0 + fr) * HDIM;
  Frag qa0, qa1;
  qa0.q[0] = *(const uint4*)(qrow + fh * 8);
  qa0.q[1] = *(const uint4*)(qrow + fh * 8 + 16);
  qa1.q[0] = *(const uint4*)(qrow + 32 + fh * 8);
  qa1.q[1] = *(const uint4*)(qrow + 32 + fh * 8 + 16);

  v8f oacc[4];
  float mrun[8], lrun[8];
#pragma unroll
  for (int c = 0; c < 4; c++)
#pragma unroll
    for (int g = 0; g < 8; g++) oacc[c][g] = 0.f;
#pragma unroll
  for (int g = 0; g < 8; g++) { mrun[g] = -1e30f; lrun[g] = 0.f; }

  int tr = tid >> 2, to = (tid & 3) * 16;  // tile row / 16-half chunk per thread

  for (int kt = 0; kt < SEQ; kt += 64) {
    __syncthreads();
    {  // K tile: [key][dim] direct async copy
      const bf16* gk = Kb + base + (size_t)(kt + tr) * HDIM + to;
      uint32_t la = lds_off(&Ks[tr * 72 + to]);
      async_b128(la, gk);
      async_b128(la + 16, gk + 8);
    }
    {  // V tile: pre-transposed in global -> direct async copy of [dim][key] rows
      const bf16* gv = Vt + base + (size_t)tr * SEQ + kt + to;
      uint32_t la = lds_off(&Vs[tr * 72 + to]);
      async_b128(la, gv);
      async_b128(la + 16, gv + 8);
    }
    wait_async0();
    __syncthreads();

    // S = Q K^T : 4 key tiles of 16
    v8f s[4];
#pragma unroll
    for (int nt = 0; nt < 4; nt++) {
#pragma unroll
      for (int g = 0; g < 8; g++) s[nt][g] = 0.f;
      v16bf b0 = load_frag_B(&Ks[(nt * 16) * 72 + 0], 72);
      v16bf b1 = load_frag_B(&Ks[(nt * 16) * 72 + 32], 72);
      s[nt] = wmma_bf16(qa0.v, b0, s[nt]);
      s[nt] = wmma_bf16(qa1.v, b1, s[nt]);
    }

    // online softmax in exp2 domain; row r = g + 8*fh lives in this 16-lane half
    float fac[8];
#pragma unroll
    for (int g = 0; g < 8; g++) {
      float mx = fmaxf(fmaxf(s[0][g], s[1][g]), fmaxf(s[2][g], s[3][g]));
      mx = red_max16(mx);
      float mnew = fmaxf(mrun[g], mx);
      fac[g] = __builtin_amdgcn_exp2f(mrun[g] - mnew);
      float rs = 0.f;
#pragma unroll
      for (int nt = 0; nt < 4; nt++) {
        float pv = __builtin_amdgcn_exp2f(s[nt][g] - mnew);
        s[nt][g] = pv;
        rs += pv;
      }
      rs = red_sum16(rs);
      lrun[g] = lrun[g] * fac[g] + rs;
      mrun[g] = mnew;
    }

    // reshape P: C-fragment -> per-wave LDS [16][64] -> two A-fragments
    bf16* Pw = Ps + wave * (16 * 72);
#pragma unroll
    for (int g = 0; g < 8; g++) {
      int row = g + fh * 8;
#pragma unroll
      for (int nt = 0; nt < 4; nt++) Pw[row * 72 + nt * 16 + fr] = (bf16)s[nt][g];
    }
    asm volatile("s_wait_dscnt 0x0" ::: "memory");
    __builtin_amdgcn_wave_barrier();
    v16bf pa0 = load_frag_A(Pw, 72);
    v16bf pa1 = load_frag_A(Pw + 32, 72);

    // rescale O, accumulate P@V
#pragma unroll
    for (int ct = 0; ct < 4; ct++)
#pragma unroll
      for (int g = 0; g < 8; g++) oacc[ct][g] *= fac[g];
#pragma unroll
    for (int ct = 0; ct < 4; ct++) {
      v16bf v0 = load_frag_B(&Vs[(ct * 16) * 72 + 0], 72);
      v16bf v1 = load_frag_B(&Vs[(ct * 16) * 72 + 32], 72);
      oacc[ct] = wmma_bf16(pa0, v0, oacc[ct]);
      oacc[ct] = wmma_bf16(pa1, v1, oacc[ct]);
    }
  }

  // epilogue: O /= l, head-concat layout [b*T+t][h*64+e]
  int b = bh >> 4, h = bh & 15;
#pragma unroll
  for (int ct = 0; ct < 4; ct++)
#pragma unroll
    for (int g = 0; g < 8; g++) {
      int row = q0 + g + fh * 8;
      int col = h * HDIM + ct * 16 + fr;
      O[((size_t)b * SEQ + row) * D_MODEL + col] = (bf16)(oacc[ct][g] / lrun[g]);
    }
}

// ---------------- host orchestration ----------------
extern "C" void kernel_launch(void* const* d_in, const int* in_sizes, int n_in,
                              void* d_out, int out_size, void* d_ws, size_t ws_size,
                              hipStream_t stream) {
  const float* x    = (const float*)d_in[0];
  const float* Wq   = (const float*)d_in[1];
  const float* Wk   = (const float*)d_in[2];
  const float* Wv   = (const float*)d_in[3];
  const float* Wo   = (const float*)d_in[4];
  const float* bo   = (const float*)d_in[5];
  const float* ln1g = (const float*)d_in[6];
  const float* ln1b = (const float*)d_in[7];
  const float* ln2g = (const float*)d_in[8];
  const float* ln2b = (const float*)d_in[9];
  const float* W1   = (const float*)d_in[10];
  const float* b1   = (const float*)d_in[11];
  const float* W2   = (const float*)d_in[12];
  const float* b2   = (const float*)d_in[13];
  float* out = (float*)d_out;

  char* ws = (char*)d_ws;
  size_t off = 0;
  auto alloc = [&](size_t bytes) -> void* {
    void* p = ws + off;
    off += (bytes + 255) & ~(size_t)255;
    return p;
  };
  bf16* h1   = (bf16*)alloc((size_t)MTOK * D_MODEL * 2);
  bf16* qkvT = (bf16*)alloc((size_t)3072 * 1024 * 2);
  bf16* WoT  = (bf16*)alloc((size_t)1024 * 1024 * 2);
  bf16* W1T  = (bf16*)alloc((size_t)1024 * 1024 * 2);
  bf16* W2T  = (bf16*)alloc((size_t)1024 * 1024 * 2);
  bf16* qbuf = (bf16*)alloc((size_t)MTOK * D_MODEL * 2);
  bf16* kbuf = (bf16*)alloc((size_t)MTOK * D_MODEL * 2);
  bf16* vbuf = (bf16*)alloc((size_t)MTOK * D_MODEL * 2);  // transposed per head
  bf16* obuf = (bf16*)alloc((size_t)MTOK * D_MODEL * 2);
  float* x1  = (float*)alloc((size_t)MTOK * D_MODEL * 4);
  bf16* h2   = (bf16*)alloc((size_t)MTOK * D_MODEL * 2);
  bf16* fbuf = (bf16*)alloc((size_t)MTOK * D_MODEL * 2);

  k_cvt_qkv<<<3072 * 1024 / 256, 256, 0, stream>>>(Wq, Wk, Wv, qkvT);
  k_cvt_T<<<1024 * 1024 / 256, 256, 0, stream>>>(Wo, WoT);
  k_cvt_T<<<1024 * 1024 / 256, 256, 0, stream>>>(W1, W1T);
  k_cvt_T<<<1024 * 1024 / 256, 256, 0, stream>>>(W2, W2T);

  k_ln<<<MTOK, 256, 0, stream>>>(x, ln1g, ln1b, h1);

  k_gemm<0><<<dim3(3072 / TN, MTOK / TM), 256, 0, stream>>>(
      h1, qkvT, nullptr, nullptr, nullptr, nullptr, qbuf, kbuf, vbuf, 3072, 1024);

  k_attn<<<dim3(SEQ / 128, BATCH * NHEADS), 256, 0, stream>>>(qbuf, kbuf, vbuf, obuf);

  k_gemm<1><<<dim3(D_MODEL / TN, MTOK / TM), 256, 0, stream>>>(
      obuf, WoT, bo, x, x1, nullptr, nullptr, nullptr, nullptr, D_MODEL, 1024);

  k_ln<<<MTOK, 256, 0, stream>>>(x1, ln2g, ln2b, h2);

  k_gemm<2><<<dim3(D_MODEL / TN, MTOK / TM), 256, 0, stream>>>(
      h2, W1T, b1, nullptr, nullptr, fbuf, nullptr, nullptr, nullptr, D_MODEL, 1024);

  k_gemm<3><<<dim3(D_MODEL / TN, MTOK / TM), 256, 0, stream>>>(
      fbuf, W2T, b2, x1, out, nullptr, nullptr, nullptr, nullptr, D_MODEL, 1024);
}
